// Greedy_Base_hGLM_55413668053222
// MI455X (gfx1250) — matrix operand order repaired and checked by reading, not compiled, not run on hardware
//
#include <hip/hip_runtime.h>
#include <hip/hip_bf16.h>

typedef _Float16 f16;
typedef __attribute__((ext_vector_type(16))) _Float16 v16h;
typedef __attribute__((ext_vector_type(8)))  _Float16 v8h;
typedef __attribute__((ext_vector_type(8)))  float    v8f;

#define SUB_NO 15
#define E_NO   64
#define I_NO   16
#define CH_NO  80          // 64 E + 16 I
#define T_NO   201         // kernel taps
#define T_DATA 300000
#define FILT_STRIDE 96     // 80 real channels + 16 zero-weighted pad (3 K-chunks of 32)

#define NBLK 8             // Toeplitz K-blocks of 32 covering taps 0..200 (+pad)
#define MACRO 256          // outputs per wave per conv tile (16x16 D tile)
#define N_TILES ((T_DATA + MACRO - 1) / MACRO)   // 1172
#define N_TTILES (T_DATA / 16)                   // 18750 (exact)
#define FILT_ROWS (N_TILES * MACRO)              // 300032: padded so K2 stores are unguarded

// ---- workspace layout (bytes) ----
#define FILT_OFF 0ull
#define FILT_BYTES (2ull * FILT_ROWS * FILT_STRIDE)              // 57,606,144
#define A_OFF   (FILT_OFF + FILT_BYTES)                          // 32B aligned
#define A_BYTES (2ull * CH_NO * NBLK * 512)                      // 655,360
#define B_OFF   (A_OFF + A_BYTES)
#define B_BYTES (2ull * 3 * 512)
#define KERN_OFF (B_OFF + B_BYTES)
#define KERN_BYTES (4ull * CH_NO * T_NO)
#define CALL_OFF (KERN_OFF + KERN_BYTES)

// Wave-level LDS fence: all LDS traffic in this pattern is produced and
// consumed by lanes of the SAME wave32; LDS ops from one wave complete in
// order, so a dscnt drain + scheduling barrier is sufficient (no s_barrier).
__device__ __forceinline__ void wave_lds_fence() {
    __builtin_amdgcn_wave_barrier();
    asm volatile("s_wait_dscnt 0" ::: "memory");
    __builtin_amdgcn_wave_barrier();
}

// =====================================================================
// K1: assignment softmax, kernel table, WMMA operand pre-packing.
// A-operand (16-bit A 16x32 layout): lane L -> M = L&15,
//   half e -> K = (L<16 ? e : 16+e)  (contiguous k order used here, same as K2's B window)
// B-operand (16-bit B 32x16 layout): lane L -> N = L&15,
//   half e -> K = (e<8 ? (L<16?0:8)+e : (L<16?16:24)+(e-8))
// =====================================================================
__global__ void hglm_prep_kernel(const float* __restrict__ temp,
                                 const float* __restrict__ u_e,
                                 const float* __restrict__ u_i,
                                 const float* __restrict__ W_syn,
                                 const float* __restrict__ Tau_syn,
                                 const float* __restrict__ Delta_syn,
                                 const float* __restrict__ Ce_logit,
                                 const float* __restrict__ Ci_logit,
                                 const int*   __restrict__ test,
                                 const int*   __restrict__ loc_e,
                                 const int*   __restrict__ loc_i,
                                 unsigned char* __restrict__ ws)
{
    float* C_all = (float*)(ws + CALL_OFF);          // [15][80]
    float* kern  = (float*)(ws + KERN_OFF);          // [80][201]
    f16*   Aws   = (f16*)(ws + A_OFF);               // [80][8][512]
    f16*   Bws   = (f16*)(ws + B_OFF);               // [3][512]
    const int tid = threadIdx.x;
    const float Tmp = temp[0];
    const int tst = test[0];

    // ---- phase 1: per-synapse-column Gumbel-softmax over 15 subunits ----
    if (tid < CH_NO) {
        const int n = tid;
        const float* logit; const float* u; int stride;
        if (n < E_NO) { logit = Ce_logit + n;        u = u_e + n;        stride = E_NO; }
        else          { logit = Ci_logit + (n-E_NO); u = u_i + (n-E_NO); stride = I_NO; }
        if (tst != 0) {
            int am = 0; float best = logit[0];
            for (int s = 1; s < SUB_NO; ++s) {
                float l = logit[s * stride];
                if (l > best) { best = l; am = s; }
            }
            for (int s = 0; s < SUB_NO; ++s) C_all[s * CH_NO + n] = (s == am) ? 1.f : 0.f;
        } else {
            float v[SUB_NO]; float mx = -1e30f;
            for (int s = 0; s < SUB_NO; ++s) {
                float uu = u[s * stride] + 1e-8f;
                float g  = -logf(-logf(uu) + 1e-8f);
                v[s] = (logit[s * stride] + g) / Tmp;
                mx = fmaxf(mx, v[s]);
            }
            float sum = 0.f;
            for (int s = 0; s < SUB_NO; ++s) { v[s] = expf(v[s] - mx); sum += v[s]; }
            const float inv = 1.f / sum;
            for (int s = 0; s < SUB_NO; ++s) C_all[s * CH_NO + n] = v[s] * inv;
        }
    }
    __syncthreads();

    // ---- phase 2: per-channel temporal kernels: tt*exp(-tt)*w ----
    for (int idx = tid; idx < CH_NO * T_NO; idx += 256) {
        const int ch = idx / T_NO, tau = idx - ch * T_NO;
        int col, loc;
        if (ch < E_NO) { col = 0; loc = loc_e[ch]; } else { col = 1; loc = loc_i[ch - E_NO]; }
        const float d  = expf(Delta_syn[loc * 2 + col]);
        const float tc = expf(Tau_syn[loc * 2 + col]);
        const float te = fmaxf((float)tau - d, 0.f);
        const float tt = te / tc;
        kern[idx] = tt * expf(-tt) * W_syn[loc * 2 + col];
    }
    __syncthreads();

    // ---- phase 3: Toeplitz A-operands, A_b[m][k] = kern[m - k + 32b] ----
    for (int idx = tid; idx < CH_NO * NBLK * 512; idx += 256) {
        const int ch = idx >> 12;
        const int r  = idx & 4095;
        const int b  = r >> 9;
        const int q  = r & 511;
        const int lane = q >> 4, e = q & 15;
        const int m = lane & 15;
        const int k = (lane < 16) ? e : 16 + e;
        const int tau = m - k + 32 * b;
        const float val = (tau >= 0 && tau < T_NO) ? kern[ch * T_NO + tau] : 0.f;
        Aws[idx] = (f16)val;
    }

    // ---- phase 4: mixing B-operand, rows 80..95 and col 15 forced to 0 ----
    for (int idx = tid; idx < 3 * 512; idx += 256) {
        const int c = idx >> 9, q = idx & 511;
        const int lane = q >> 4, e = q & 15;
        const int s = lane & 15;
        const int k = (e < 8) ? ((lane < 16 ? 0 : 8) + e)
                              : ((lane < 16 ? 16 : 24) + (e - 8));
        const int ch = 32 * c + k;
        const float val = (s < SUB_NO && ch < CH_NO) ? C_all[s * CH_NO + ch] : 0.f;
        Bws[idx] = (f16)val;
    }
}

// =====================================================================
// K2: depthwise causal conv via Toeplitz-blocked WMMA.
// One wave = one channel x one 256-output macro-tile.
// D[m][j] = filt[t0 + 16j + m], contraction over 8 blocks of 32 lags.
// =====================================================================
__global__ void hglm_conv_kernel(const float* __restrict__ Se,
                                 const float* __restrict__ Si,
                                 unsigned char* __restrict__ ws)
{
    __shared__ f16 win[8][512];                   // per-wave signal window (f16)
    const f16* Aws  = (const f16*)(ws + A_OFF);
    f16*       filt = (f16*)(ws + FILT_OFF);

    const int wave = threadIdx.x >> 5;
    const int lane = threadIdx.x & 31;
    const int ch   = blockIdx.y * 8 + wave;       // 0..79
    const int t0   = blockIdx.x * MACRO;

    // preload this channel's 8 Toeplitz A-operands (one v16h per lane each)
    v16h Ab[NBLK];
#pragma unroll
    for (int b = 0; b < NBLK; ++b)
        Ab[b] = *(const v16h*)(Aws + ((ch * NBLK + b) << 9) + (lane << 4));

    // stage the [t0-224, t0+271] window of this channel into LDS as f16,
    // pair-packed so DS traffic is b32 stores
    const float* src; int stride, cofs;
    if (ch < E_NO) { src = Se; stride = E_NO; cofs = ch; }
    else           { src = Si; stride = I_NO; cofs = ch - E_NO; }
    const int base_t = t0 - 224 + lane * 16;
    unsigned* win32 = (unsigned*)&win[wave][0];
#pragma unroll
    for (int i = 0; i < 8; ++i) {
        const int ta = base_t + 2 * i;
        const int tb = ta + 1;
        const float va = (ta >= 0 && ta < T_DATA) ? src[ta * stride + cofs] : 0.f;
        const float vb = (tb >= 0 && tb < T_DATA) ? src[tb * stride + cofs] : 0.f;
        union { f16 h[2]; unsigned u; } p;
        p.h[0] = (f16)va; p.h[1] = (f16)vb;
        win32[lane * 8 + i] = p.u;
    }
    wave_lds_fence();

    v8f acc;
#pragma unroll
    for (int i = 0; i < 8; ++i) acc[i] = 0.f;

    const int j    = lane & 15;
    const int koff = (lane < 16) ? 0 : 16;
#pragma unroll
    for (int b = 0; b < NBLK; ++b) {
        const v16h bv = *(const v16h*)&win[wave][16 * j + koff + 224 - 32 * b];
        acc = __builtin_amdgcn_wmma_f32_16x16x32_f16(false, Ab[b], false, bv,
                                                     (short)0, acc, false, false);
    }

    // filt region is padded to FILT_ROWS (>= t0+255 for every tile): no guards
    const int mb = (lane < 16) ? 0 : 8;
    f16* dst = filt + (t0 + 16 * j + mb) * FILT_STRIDE + ch;
#pragma unroll
    for (int v = 0; v < 8; ++v)
        dst[v * FILT_STRIDE] = (f16)acc[v];
}

// =====================================================================
// K3: mixing GEMM (filt x C^T via 3 WMMAs, K-chunks of 32 channels)
// + per-timestep tanh dendrite-tree recursion driven by C_den.
// One wave = one 16-timestep tile.
// =====================================================================
__global__ void hglm_mix_tree_kernel(const float* __restrict__ W_sub,
                                     const float* __restrict__ V_o,
                                     const int*   __restrict__ C_den,
                                     float* __restrict__ out,
                                     unsigned char* __restrict__ ws)
{
    __shared__ float synlds[8][256];              // per-wave 16x16 D transpose
    const f16* filt = (const f16*)(ws + FILT_OFF);
    const f16* Bws  = (const f16*)(ws + B_OFF);

    const int wave = threadIdx.x >> 5;
    const int lane = threadIdx.x & 31;
    int g = blockIdx.x * 8 + wave;
    if (g >= N_TTILES) g = N_TTILES - 1;          // duplicate tail work, same data
    const int t0 = g * 16;

    const int t  = t0 + (lane & 15);              // A row (time) for this lane
    const int kb = (lane < 16) ? 0 : 8;
    const f16* row = filt + t * FILT_STRIDE;

    v8f acc;
#pragma unroll
    for (int i = 0; i < 8; ++i) acc[i] = 0.f;

#pragma unroll
    for (int c = 0; c < 3; ++c) {
        const v8h lo = *(const v8h*)(row + 32 * c + kb);
        const v8h hi = *(const v8h*)(row + 32 * c + 16 + kb);
        v16h a;
#pragma unroll
        for (int i = 0; i < 8; ++i) { a[i] = lo[i]; a[i + 8] = hi[i]; }
        const v16h bv = *(const v16h*)(Bws + (c << 9) + (lane << 4));
        acc = __builtin_amdgcn_wmma_f32_16x16x32_f16(false, a, false, bv,
                                                     (short)0, acc, false, false);
    }

    // transpose D through LDS: lane holds subunit s = lane&15, times mb..mb+7
    const int s  = lane & 15;
    const int mb = (lane < 16) ? 0 : 8;
#pragma unroll
    for (int v = 0; v < 8; ++v) synlds[wave][s * 16 + mb + v] = acc[v];
    wave_lds_fence();

    if (lane < 16) {
        const int m = lane;
        float x[SUB_NO], c[SUB_NO], W[SUB_NO];
#pragma unroll
        for (int si = 0; si < SUB_NO; ++si) x[si] = synlds[wave][si * 16 + m];
#pragma unroll
        for (int si = 0; si < SUB_NO; ++si) W[si] = W_sub[si];
        // leaves-first recursion: children always have larger indices
#pragma unroll
        for (int si = SUB_NO - 1; si >= 0; --si) {
            float a2 = x[si];
#pragma unroll
            for (int jj = si + 1; jj < SUB_NO; ++jj)
                if (C_den[si * SUB_NO + jj] != 0) a2 += c[jj] * W[jj];
            c[si] = tanhf(a2);
        }
        out[t0 + m] = c[0] * W[0] + V_o[0];
    }
}

extern "C" void kernel_launch(void* const* d_in, const int* in_sizes, int n_in,
                              void* d_out, int out_size, void* d_ws, size_t ws_size,
                              hipStream_t stream) {
    const float* Se      = (const float*)d_in[0];
    const float* Si      = (const float*)d_in[1];
    const float* temp    = (const float*)d_in[2];
    const float* u_e     = (const float*)d_in[3];
    const float* u_i     = (const float*)d_in[4];
    const float* W_syn   = (const float*)d_in[5];
    const float* Tau_syn = (const float*)d_in[6];
    const float* Delta   = (const float*)d_in[7];
    const float* W_sub   = (const float*)d_in[8];
    const float* V_o     = (const float*)d_in[9];
    const float* Ce_log  = (const float*)d_in[10];
    const float* Ci_log  = (const float*)d_in[11];
    const int*   test    = (const int*)d_in[12];
    const int*   C_den   = (const int*)d_in[13];
    const int*   loc_e   = (const int*)d_in[14];
    const int*   loc_i   = (const int*)d_in[15];
    unsigned char* ws    = (unsigned char*)d_ws;
    float* out           = (float*)d_out;

    hglm_prep_kernel<<<1, 256, 0, stream>>>(temp, u_e, u_i, W_syn, Tau_syn, Delta,
                                            Ce_log, Ci_log, test, loc_e, loc_i, ws);

    dim3 gconv(N_TILES, 10, 1);                   // 1172 time macro-tiles x 80 ch
    hglm_conv_kernel<<<gconv, 256, 0, stream>>>(Se, Si, ws);

    const int nblk3 = (N_TTILES + 7) / 8;         // 2344 blocks, 8 waves each
    hglm_mix_tree_kernel<<<nblk3, 256, 0, stream>>>(W_sub, V_o, C_den, out, ws);
}
